// MultiHeadAttention_44306882625664
// MI455X (gfx1250) — matrix operand order repaired and checked by reading, not compiled, not run on hardware
//
#include <hip/hip_runtime.h>

#define DEV __device__ __forceinline__

// ---- constants for this problem ----
#define NB   4
#define NH   16
#define DEP  64
#define LL   2048
#define DM   1024
#define MM   (NB * LL)          // 8192 rows of the projection GEMMs

typedef __attribute__((ext_vector_type(16))) __bf16 bf16x16;
typedef __attribute__((ext_vector_type(8)))  __bf16 bf16x8;
typedef __attribute__((ext_vector_type(8)))  float  f32x8;
typedef __attribute__((ext_vector_type(4)))  float  f32x4;

// ---------------- WMMA helper (v_wmma_f32_16x16x32_bf16) ----------------
DEV f32x8 wmma_bf16(bf16x16 a, bf16x16 b, f32x8 c) {
    return __builtin_amdgcn_wmma_f32_16x16x32_bf16(
        /*neg_a=*/false, a, /*neg_b=*/false, b,
        /*c_mod=*/(short)0, c, /*reuse_a=*/false, /*reuse_b=*/false);
}

// A fragment, 16x32 bf16, row-major source with row stride lda (elements).
// Lane l: row = l&15, K = [8h..8h+7] then [16+8h..16+8h+7], h = l>>4.
DEV bf16x16 load_a_frag(const __bf16* rowbase, int lda, int k0, int lane) {
    int m = lane & 15, h = (lane >> 4) & 1;
    const __bf16* p = rowbase + (size_t)m * lda + k0 + 8 * h;
    bf16x16 r;
    ((bf16x8*)&r)[0] = *((const bf16x8*)p);
    ((bf16x8*)&r)[1] = *((const bf16x8*)(p + 16));
    return r;
}

// A fragment from fp32 source (generic pointer: global or LDS), cvt to bf16.
DEV bf16x16 load_a_frag(const float* rowbase, int lda, int k0, int lane) {
    int m = lane & 15, h = (lane >> 4) & 1;
    const float* p = rowbase + (size_t)m * lda + k0 + 8 * h;
    f32x4 x0 = *((const f32x4*)(p));
    f32x4 x1 = *((const f32x4*)(p + 4));
    f32x4 x2 = *((const f32x4*)(p + 16));
    f32x4 x3 = *((const f32x4*)(p + 20));
    bf16x16 r;
#pragma unroll
    for (int i = 0; i < 4; ++i) {
        r[i]      = (__bf16)x0[i];
        r[i + 4]  = (__bf16)x1[i];
        r[i + 8]  = (__bf16)x2[i];
        r[i + 12] = (__bf16)x3[i];
    }
    return r;
}

// B fragment, 32x16 bf16, where B[k][n] = S[n][k] (S row-major, stride ldb).
// Lane l: col n = l&15, K = [8h..8h+7] then [16+8h..16+8h+7], h = l>>4.
DEV bf16x16 load_b_frag(const __bf16* colbase, int ldb, int k0, int lane) {
    int n = lane & 15, h = (lane >> 4) & 1;
    const __bf16* p = colbase + (size_t)n * ldb + k0 + 8 * h;
    bf16x16 r;
    ((bf16x8*)&r)[0] = *((const bf16x8*)p);
    ((bf16x8*)&r)[1] = *((const bf16x8*)(p + 16));
    return r;
}

// B fragment from fp32 source (weight matrices), converted to bf16.
DEV bf16x16 load_b_frag(const float* colbase, int ldb, int k0, int lane) {
    int n = lane & 15, h = (lane >> 4) & 1;
    const float* p = colbase + (size_t)n * ldb + k0 + 8 * h;
    bf16x16 r;
#pragma unroll
    for (int j = 0; j < 4; ++j) {
        f32x4 x = *((const f32x4*)(p + 4 * j + 8 * (j >> 1)));
#pragma unroll
        for (int i = 0; i < 4; ++i) r[4 * j + i] = (__bf16)x[i];
    }
    return r;
}

// ---------------- Kernel 1/4: Out = A @ W^T + bias ----------------
// A: [M,K] (fp32 or bf16), W: [Nn,K] fp32, Out: [M,Nn] (bf16 or fp32).
// One 16x16 C tile per wave, 8 waves per block.
template <typename AT, typename OT>
__global__ __launch_bounds__(256) void gemm_xwT(const AT* __restrict__ A,
                                                const float* __restrict__ W,
                                                const float* __restrict__ bias,
                                                OT* __restrict__ Out,
                                                int M, int K, int Nn) {
    int lane = threadIdx.x & 31;
    int wave = threadIdx.x >> 5;
    int tile = blockIdx.x * 8 + wave;
    int ntn  = Nn >> 4;
    if (tile >= (M >> 4) * ntn) return;
    int mt = tile / ntn, nt = tile % ntn;

    const AT*    Arow = A + (size_t)(mt * 16) * K;
    const float* Wcol = W + (size_t)(nt * 16) * K;

    f32x8 acc = {};
    for (int k0 = 0; k0 < K; k0 += 32) {
        bf16x16 af = load_a_frag(Arow, K, k0, lane);
        bf16x16 bf = load_b_frag(Wcol, K, k0, lane);
        acc = wmma_bf16(af, bf, acc);
    }
    int n = lane & 15, hh = (lane >> 4) & 1;
    float bv = bias[nt * 16 + n];
#pragma unroll
    for (int r = 0; r < 8; ++r) {
        int gm = mt * 16 + 8 * hh + r;
        Out[(size_t)gm * Nn + nt * 16 + n] = (OT)(acc[r] + bv);
    }
}

// ---------------- Kernel 2: qhT[nh][d][l] = qp[n][l][h*64+d] (bf16) -------
__global__ __launch_bounds__(256) void transpose_qh(const __bf16* __restrict__ qp,
                                                    __bf16* __restrict__ qhT) {
    __shared__ __bf16 tile[64][65];
    int b  = blockIdx.x;            // NB*NH * (LL/64) = 64*32
    int lt = b & 31;
    int nh = b >> 5;
    int n = nh >> 4, h = nh & 15;
    int lbase = lt * 64;
    int t = threadIdx.x;
#pragma unroll
    for (int i = 0; i < 16; ++i) {
        int idx = t + i * 256;
        int li = idx >> 6, di = idx & 63;
        tile[li][di] = qp[(size_t)(n * LL + lbase + li) * DM + h * DEP + di];
    }
    __syncthreads();
#pragma unroll
    for (int i = 0; i < 16; ++i) {
        int idx = t + i * 256;
        int di = idx >> 6, li = idx & 63;
        qhT[(size_t)(nh * DEP + di) * LL + lbase + li] = tile[li][di];
    }
}

// ------- Kernel 3: fused scores + mask + softmax + ctx GEMM ---------------
// One block = one (n,h) x 16-row strip. Full 16x2048 fp32 strip in LDS.
// Phase 1: scores strip via WMMA (each wave: 256 columns).
// Phase 2: coalesced mask add + softmax per row; attn written to global AND LDS.
// Phase 3: ctx strip = attn_strip @ qh using LDS A-frags and L2-resident qhT.
__global__ __launch_bounds__(256) void scores_softmax_ctx(
        const __bf16* __restrict__ qp, const float* __restrict__ mask,
        const __bf16* __restrict__ qhT, float* __restrict__ attn,
        __bf16* __restrict__ ctx) {
    extern __shared__ float s[];          // 16 rows x SROW floats (+ reuse)
    const int SROW = LL + 16;
    int lane = threadIdx.x & 31;
    int wave = threadIdx.x >> 5;
    int b  = blockIdx.x;                  // NB*NH*(LL/16) = 64*128
    int mt = b & 127;
    int nh = b >> 7;
    int n = nh >> 4, h = nh & 15;
    int mbase = mt * 16;

    // ---- Phase 1: raw scores (qh @ qh^T * 1/8) into LDS strip ----
    const __bf16* qbase = qp + (size_t)n * LL * DM + h * DEP;
    const __bf16* Arow  = qbase + (size_t)mbase * DM;
    bf16x16 a0 = load_a_frag(Arow, DM, 0, lane);
    bf16x16 a1 = load_a_frag(Arow, DM, 32, lane);

    int nn = lane & 15, hh = (lane >> 4) & 1;
    for (int ct = 0; ct < 16; ++ct) {     // each wave covers 256 columns
        int nbase = wave * 256 + ct * 16;
        const __bf16* Bcol = qbase + (size_t)nbase * DM;
        bf16x16 b0 = load_b_frag(Bcol, DM, 0, lane);
        bf16x16 b1 = load_b_frag(Bcol, DM, 32, lane);
        f32x8 acc = {};
        acc = wmma_bf16(a0, b0, acc);
        acc = wmma_bf16(a1, b1, acc);
#pragma unroll
        for (int r = 0; r < 8; ++r) {
            int lr = 8 * hh + r;
            s[lr * SROW + nbase + nn] = acc[r] * 0.125f;
        }
    }
    __syncthreads();

    // ---- Phase 2: coalesced mask add + softmax (each wave owns 2 rows) ----
#pragma unroll
    for (int rr = 0; rr < 2; ++rr) {
        int lr = wave * 2 + rr;
        float* row = s + lr * SROW;
        const float* mrow = mask + ((size_t)nh * LL + mbase + lr) * LL;
        float mx = -3.0e38f;
        for (int c = lane; c < LL; c += 32) {     // 128B/wave coalesced mask read
            float x = row[c] + mrow[c] * (-1e9f);
            row[c] = x;
            mx = fmaxf(mx, x);
        }
#pragma unroll
        for (int off = 16; off; off >>= 1) mx = fmaxf(mx, __shfl_xor(mx, off, 32));
        float sum = 0.f;
        for (int c = lane; c < LL; c += 32) {
            float e = __expf(row[c] - mx);
            row[c] = e;
            sum += e;
        }
#pragma unroll
        for (int off = 16; off; off >>= 1) sum += __shfl_xor(sum, off, 32);
        float inv = 1.0f / sum;
        float* orow = attn + ((size_t)nh * LL + mbase + lr) * LL;
        for (int c = lane; c < LL; c += 32) {
            float v = row[c] * inv;
            row[c] = v;                            // keep normalized attn in LDS
            orow[c] = v;                           // coalesced global attn write
        }
    }
    __syncthreads();

    // ---- Phase 3: ctx strip = attn_strip @ qh ----
    // wave w: d-tile nt = w&3, K half kh = w>>2 (waves 0-3: K[0,1024), 4-7: rest)
    int nt = wave & 3;
    int kh = wave >> 2;
    int kbase = kh * 1024;
    const __bf16* Bcol = qhT + (size_t)(nh * DEP + nt * 16) * LL;
    f32x8 acc = {};
    for (int k0 = 0; k0 < 1024; k0 += 32) {
        bf16x16 af = load_a_frag((const float*)s, SROW, kbase + k0, lane); // LDS
        bf16x16 bf = load_b_frag(Bcol, LL, kbase + k0, lane);              // L2
        acc = wmma_bf16(af, bf, acc);
    }
    __syncthreads();                       // strip no longer needed; reuse as scratch
    if (wave >= 4) {                       // upper-K partials -> LDS scratch
        float* part = s + ((size_t)(wave - 4) * 32 + lane) * 8;
#pragma unroll
        for (int r = 0; r < 8; ++r) part[r] = acc[r];
    }
    __syncthreads();
    if (wave < 4) {                        // combine halves, store ctx (bf16)
        const float* part = s + ((size_t)wave * 32 + lane) * 8;
#pragma unroll
        for (int r = 0; r < 8; ++r) {
            int gm = mbase + 8 * hh + r;
            float v = acc[r] + part[r];
            ctx[(size_t)(n * LL + gm) * DM + h * DEP + nt * 16 + nn] = (__bf16)v;
        }
    }
}

// ---------------------------- launcher ------------------------------------
extern "C" void kernel_launch(void* const* d_in, const int* in_sizes, int n_in,
                              void* d_out, int out_size, void* d_ws, size_t ws_size,
                              hipStream_t stream) {
    const float* q    = (const float*)d_in[0];
    // d_in[1]=k, d_in[2]=v are dead per the reference's faithful bug
    const float* mask = (const float*)d_in[3];
    const float* wq_w = (const float*)d_in[4];
    const float* wq_b = (const float*)d_in[5];
    const float* wo_w = (const float*)d_in[10];
    const float* wo_b = (const float*)d_in[11];

    float* out  = (float*)d_out;                         // [N,L,DM]
    float* attn = out + (size_t)NB * LL * DM;            // [N,H,L,L]

    char* ws = (char*)d_ws;
    __bf16* qp  = (__bf16*)(ws);                          // 16 MiB
    __bf16* qhT = (__bf16*)(ws + (size_t)16 * 1024 * 1024);
    __bf16* ctx = (__bf16*)(ws + (size_t)32 * 1024 * 1024);

    // 1) qp = q @ Wq^T + b   (fp32 in, bf16 out)
    gemm_xwT<float, __bf16><<<4096, 256, 0, stream>>>(q, wq_w, wq_b, qp, MM, DM, DM);
    // 2) head-transpose qh -> qhT (bf16)
    transpose_qh<<<NB * NH * (LL / 64), 256, 0, stream>>>(qp, qhT);
    // 3) fused scores + mask + softmax + ctx (attn is second output)
    scores_softmax_ctx<<<NB * NH * (LL / 16), 256,
                         (size_t)16 * (LL + 16) * sizeof(float), stream>>>(
        qp, mask, qhT, attn, ctx);
    // 4) out = ctx @ Wo^T + b  (bf16 in, fp32 out)
    gemm_xwT<__bf16, float><<<4096, 256, 0, stream>>>(ctx, wo_w, wo_b, out, MM, DM, DM);
}
